// GCN_55181739819641
// MI455X (gfx1250) — matrix operand order repaired and checked by reading, not compiled, not run on hardware
//
#include <hip/hip_runtime.h>
#include <hip/hip_bf16.h>

typedef __attribute__((ext_vector_type(16))) __bf16 v16bf;
typedef __attribute__((ext_vector_type(8)))  __bf16 v8bf;
typedef __attribute__((ext_vector_type(8)))  float  v8f;
typedef __attribute__((ext_vector_type(4)))  unsigned int u32x4;
typedef __attribute__((ext_vector_type(8)))  int  i32x8;
typedef __attribute__((ext_vector_type(4)))  int  i32x4;

#define FEAT 128   // d_in == d_hid == d_out == 128

static __device__ __forceinline__ unsigned short f32_to_bf16_rne(float f) {
    unsigned u = __builtin_bit_cast(unsigned, f);
    u += 0x7FFFu + ((u >> 16) & 1u);          // round to nearest even
    return (unsigned short)(u >> 16);
}

// ---------------------------------------------------------------------------
// Degree count at edge targets (col), int64 edge indices.
__global__ __launch_bounds__(256) void deg_kernel(const long long* __restrict__ col,
                                                  unsigned* __restrict__ cnt, int E) {
    int e = blockIdx.x * 256 + threadIdx.x;
    if (e < E) atomicAdd(&cnt[(int)col[e]], 1u);
}

// dinv[i] = rsqrt(deg[i] + 1)   (self loop)
__global__ __launch_bounds__(256) void dinv_kernel(const unsigned* __restrict__ cnt,
                                                   float* __restrict__ dinv, int n) {
    int i = blockIdx.x * 256 + threadIdx.x;
    if (i < n) dinv[i] = rsqrtf((float)cnt[i] + 1.0f);
}

// ---------------------------------------------------------------------------
// f32 -> bf16 (4 elements / thread)
__global__ __launch_bounds__(256) void cvt_bf16_kernel(const float* __restrict__ in,
                                                       __bf16* __restrict__ out, int n4) {
    int i = blockIdx.x * 256 + threadIdx.x;
    if (i >= n4) return;
    float4 v = ((const float4*)in)[i];
    uint2 pk;
    pk.x = (unsigned)f32_to_bf16_rne(v.x) | ((unsigned)f32_to_bf16_rne(v.y) << 16);
    pk.y = (unsigned)f32_to_bf16_rne(v.z) | ((unsigned)f32_to_bf16_rne(v.w) << 16);
    ((uint2*)out)[i] = pk;
}

// ---------------------------------------------------------------------------
// Pack W [128x128 f32, row-major K x N] into per-lane WMMA B-fragment order.
// Flat index: (((kt*8 + nt)*32 + lane)*16 + j), lane holds col = nt*16+(lane&15),
// K = kt*32 + (lane<16 ? 0 : 8) + (j<8 ? j : j+8).
__global__ __launch_bounds__(256) void pack_w_kernel(const float* __restrict__ W,
                                                     __bf16* __restrict__ wp) {
    int t = blockIdx.x * 256 + threadIdx.x;   // 0..1023
    if (t >= 1024) return;
    int lane = t & 31;
    int nt   = (t >> 5) & 7;
    int kt   = t >> 8;
    int col  = nt * 16 + (lane & 15);
    int kb   = (lane < 16) ? 0 : 8;
    __bf16* dst = wp + (((kt * 8 + nt) * 32 + lane) * 16);
#pragma unroll
    for (int j = 0; j < 16; ++j) {
        int K = kt * 32 + kb + (j < 8 ? j : j + 8);
        unsigned short s = f32_to_bf16_rne(W[K * FEAT + col]);
        dst[j] = __builtin_bit_cast(__bf16, s);
    }
}

// ---------------------------------------------------------------------------
// h = xb @ W  via v_wmma_f32_16x16x32_bf16, with the packed 32KB weight block
// staged into LDS once per workgroup by the Tensor Data Mover.
#define WP_BYTES 32768          // 128*128 bf16
#define WP_ELTS4 8192           // as 4-byte elements

__global__ __launch_bounds__(256) void gemm_kernel(const __bf16* __restrict__ xb,
                                                   const __bf16* __restrict__ wp,
                                                   float* __restrict__ h, int mtiles) {
    __shared__ __align__(16) char lds_wp[WP_BYTES];

    const int wave = threadIdx.x >> 5;
    const int lane = threadIdx.x & 31;

    if (wave == 0) {
        // ---- TDM: DMA 32KB weight block (global -> LDS), one descriptor ----
        unsigned lds_off = (unsigned)(size_t)(void*)lds_wp;   // low 32b = LDS offset
        unsigned long long ga = (unsigned long long)(const void*)wp;
        // D# group 0: count=1 | lds_addr | global_addr[56:0] | type=2
        u32x4 g0;
        g0[0] = 1u;                                    // count=1, user, no gather
        g0[1] = lds_off;                               // lds_addr (bytes)
        g0[2] = (unsigned)(ga & 0xFFFFFFFFu);          // global_addr[31:0]
        g0[3] = (unsigned)((ga >> 32) & 0x01FFFFFFu)   // global_addr[56:32]
              | (2u << 30);                            // type = 2 ("image")
        // D# group 1: data_size=4B, 1 row of 8192 elements
        i32x8 g1;
        g1[0] = (2 << 16);                             // data_size = 2 (4 bytes)
        g1[1] = (int)((unsigned)WP_ELTS4 << 16);       // tensor_dim0[15:0] @ bits 63:48
        g1[2] = (int)(1u << 16);                       // tensor_dim0[31:16]=0, tensor_dim1=1
        g1[3] = (int)((unsigned)WP_ELTS4 << 16);       // tile_dim0 @ bits 127:112
        g1[4] = 1;                                     // tile_dim1=1, tile_dim2=0
        g1[5] = WP_ELTS4;                              // tensor_dim0_stride[31:0]
        g1[6] = 0;
        g1[7] = 0;
        i32x4 z4 = {0, 0, 0, 0};                       // groups 2/3 unused (<=2D)
        i32x8 z8 = {0, 0, 0, 0, 0, 0, 0, 0};           // trailing group (6-arg form)
        __builtin_amdgcn_tensor_load_to_lds(g0, g1, z4, z4, z8, 0);
        __builtin_amdgcn_s_wait_tensorcnt(0);
    }
    __syncthreads();                                   // LDS weights visible to all waves

    const int mtile = blockIdx.x * 8 + wave;
    if (mtile < mtiles) {                              // wave-uniform: EXEC all-ones inside
        const int row0 = mtile * 16;
        const int arow = row0 + (lane & 15);
        const int kb   = (lane < 16) ? 0 : 8;          // A-matrix 16-bit lane K-base

        v8f acc[8];
#pragma unroll
        for (int nt = 0; nt < 8; ++nt) acc[nt] = (v8f){};

#pragma unroll
        for (int kt = 0; kt < 4; ++kt) {
            // A fragment: two contiguous 16B chunks (K = kt*32+kb+{0..7,16..23})
            const v8bf* pa = (const v8bf*)(xb + (size_t)arow * FEAT + kt * 32 + kb);
            v8bf alo = pa[0];
            v8bf ahi = pa[2];                          // +16 bf16 elements
            v16bf a = __builtin_shufflevector(alo, ahi,
                        0, 1, 2, 3, 4, 5, 6, 7, 8, 9, 10, 11, 12, 13, 14, 15);
            const v16bf* pb = (const v16bf*)lds_wp + (kt * 8) * 32 + lane;
#pragma unroll
            for (int nt = 0; nt < 8; ++nt) {
                v16bf b = pb[nt * 32];                 // ds_load from staged weights
                acc[nt] = __builtin_amdgcn_wmma_f32_16x16x32_bf16(
                    false, a, false, b, (short)0, acc[nt], false, false);
            }
        }

        // C/D layout: lane = n + (m>=8 ? 16 : 0), vgpr r -> row (lane<16?0:8)+r
        const int n     = lane & 15;
        const int rbase = (lane < 16) ? 0 : 8;
#pragma unroll
        for (int nt = 0; nt < 8; ++nt)
#pragma unroll
            for (int r = 0; r < 8; ++r)
                h[(size_t)(row0 + rbase + r) * FEAT + nt * 16 + n] = acc[nt][r];
    }
}

// ---------------------------------------------------------------------------
// agg[col] += (dinv[row]*dinv[col]) * h[row]   — one wave per edge,
// lane covers 4 features (float4 load + 4 f32 atomics).
__global__ __launch_bounds__(256) void scatter_kernel(const long long* __restrict__ erow,
                                                      const long long* __restrict__ ecol,
                                                      const float* __restrict__ dinv,
                                                      const float* __restrict__ h,
                                                      float* __restrict__ agg, int E) {
    int wid  = (blockIdx.x * 256 + threadIdx.x) >> 5;
    int lane = threadIdx.x & 31;
    if (wid >= E) return;
    int r = (int)erow[wid];
    int c = (int)ecol[wid];
    float nrm = dinv[r] * dinv[c];
    float4 hv = ((const float4*)(h + (size_t)r * FEAT))[lane];
    float* dst = agg + (size_t)c * FEAT + lane * 4;
    atomicAdd(dst + 0, nrm * hv.x);
    atomicAdd(dst + 1, nrm * hv.y);
    atomicAdd(dst + 2, nrm * hv.z);
    atomicAdd(dst + 3, nrm * hv.w);
}

// ---------------------------------------------------------------------------
// out = [relu]( agg + dinv^2 * h + bias )   — float4 per thread
__global__ __launch_bounds__(256) void finalize_kernel(const float* __restrict__ agg,
                                                       const float* __restrict__ h,
                                                       const float* __restrict__ dinv,
                                                       const float* __restrict__ bias,
                                                       float* __restrict__ out,
                                                       int n4, int relu) {
    int i = blockIdx.x * 256 + threadIdx.x;
    if (i >= n4) return;
    int node = i >> 5;                    // 32 float4 per node
    int fb   = (i & 31);
    float di = dinv[node];
    float d2 = di * di;
    float4 a = ((const float4*)agg)[i];
    float4 hv = ((const float4*)h)[i];
    float4 b = ((const float4*)bias)[fb];
    float4 v;
    v.x = a.x + d2 * hv.x + b.x;
    v.y = a.y + d2 * hv.y + b.y;
    v.z = a.z + d2 * hv.z + b.z;
    v.w = a.w + d2 * hv.w + b.w;
    if (relu) {
        v.x = fmaxf(v.x, 0.f); v.y = fmaxf(v.y, 0.f);
        v.z = fmaxf(v.z, 0.f); v.w = fmaxf(v.w, 0.f);
    }
    ((float4*)out)[i] = v;
}

// ---------------------------------------------------------------------------
extern "C" void kernel_launch(void* const* d_in, const int* in_sizes, int n_in,
                              void* d_out, int out_size, void* d_ws, size_t ws_size,
                              hipStream_t stream) {
    const float*     x    = (const float*)d_in[0];
    const long long* eidx = (const long long*)d_in[1];     // int64 (2,E) flat
    const float* W[3] = { (const float*)d_in[2], (const float*)d_in[4], (const float*)d_in[6] };
    const float* B[3] = { (const float*)d_in[3], (const float*)d_in[5], (const float*)d_in[7] };

    const int N = in_sizes[0] / FEAT;          // 50000
    const int E = in_sizes[1] / 2;             // 625000
    const long long NE = (long long)N * FEAT;  // feature elements
    const int n4 = (int)(NE / 4);
    const int mtiles = N / 16;                 // 3125 (N divisible by 16)

    // Workspace carve-up (~64.5 MB)
    char* ws = (char*)d_ws;
    size_t szF = (size_t)NE * sizeof(float);   // 25.6 MB
    float*    hbuf = (float*)ws;                       // GEMM output
    float*    abuf = (float*)(ws + szF);               // aggregation / layer output
    __bf16*   xb   = (__bf16*)(ws + 2 * szF);          // bf16 activations
    float*    dinv = (float*)(ws + 2 * szF + (size_t)NE * 2);
    unsigned* cnt  = (unsigned*)((char*)dinv + (size_t)N * 4);
    __bf16*   wp   = (__bf16*)((char*)cnt + (size_t)N * 4);

    const long long* erow = eidx;
    const long long* ecol = eidx + E;

    // --- degrees / dinv (layer-invariant) ---
    (void)hipMemsetAsync(cnt, 0, (size_t)N * 4, stream);
    deg_kernel<<<(E + 255) / 256, 256, 0, stream>>>(ecol, cnt, E);
    dinv_kernel<<<(N + 255) / 256, 256, 0, stream>>>(cnt, dinv, N);

    const float* layer_in = x;
    for (int L = 0; L < 3; ++L) {
        float* layer_out = (L == 2) ? (float*)d_out : abuf;
        // activations -> bf16
        cvt_bf16_kernel<<<(n4 + 255) / 256, 256, 0, stream>>>(layer_in, xb, n4);
        // pack weights into WMMA B-fragment order
        pack_w_kernel<<<4, 256, 0, stream>>>(W[L], wp);
        // h = xb @ W  (TDM-staged weights + WMMA bf16 -> f32)
        gemm_kernel<<<(mtiles + 7) / 8, 256, 0, stream>>>(xb, wp, hbuf, mtiles);
        // neighbor aggregation
        (void)hipMemsetAsync(abuf, 0, szF, stream);
        scatter_kernel<<<(E + 7) / 8, 256, 0, stream>>>(erow, ecol, dinv, hbuf, abuf, E);
        // + self loop + bias (+ relu on layers 0,1)
        finalize_kernel<<<(n4 + 255) / 256, 256, 0, stream>>>(
            abuf, hbuf, dinv, B[L], layer_out, n4, (L < 2) ? 1 : 0);
        layer_in = layer_out;
    }
}
